// Context_Aware_Att_60284160967214
// MI455X (gfx1250) — matrix operand order repaired and checked by reading, not compiled, not run on hardware
//
#include <hip/hip_runtime.h>
#include <hip/hip_bf16.h>

// Context-aware attention for MI455X (gfx1250, wave32, WMMA f16 16x16x32).
//
// Pipeline:
//   f32_to_f16 x3  : Wq/Wk/Wv -> f16 copies (so GEMM B-fragments are single
//                    32B loads with no in-loop convert chain)
//   proj_kernel x3 : f32 activations -> f16 Qh [B,32,1024], Kh [B,160,1024],
//                    VhT [B,H,64,160] (V transposed per head: attention
//                    B-fragments become contiguous 32B loads)
//   attn_kernel    : per (b,h): S=Qh*Kh^T/8, E=exp(S)*mask, denom=rowsum(E),
//                    ctx = (E @ Vh) / (denom + 1e-8)   (all GEMMs via WMMA)
//
// Workspace: Qh 4,194,304 + Kh 20,971,520 + VhT 20,971,520 + 3x1,048,576 f16
//            = 49,283,072 f16 = 98,566,144 bytes.

typedef __attribute__((ext_vector_type(16))) _Float16 v16h;
typedef __attribute__((ext_vector_type(8)))  _Float16 v8h;
typedef __attribute__((ext_vector_type(8)))  float    v8f;
typedef __attribute__((ext_vector_type(4)))  float    f32x4;

#define DMODEL 1024   // input model dim (also H*DK)
#define NHEADS 16
#define DKH    64
#define NLQ    32
#define NLK    128
#define NLT    160    // LK + LQ
#define NB     128

// ---- fragment builders -----------------------------------------------------

// A-matrix 16x32 f16 fragment from f32 memory. Lane layout used here (contiguous k, B uses the same order):
// lane holds row lane%16; values K = {c0..c0+15}, c0=16*(lane/16).
// `p` must already point at (row, kbase + c0).
static __device__ __forceinline__ v16h a_frag_from_f32(const float* p) {
  f32x4 x0 = *(const f32x4*)(p + 0);
  f32x4 x1 = *(const f32x4*)(p + 4);
  f32x4 x2 = *(const f32x4*)(p + 8);
  f32x4 x3 = *(const f32x4*)(p + 12);
  v16h r;
  r[0]=(_Float16)x0[0];  r[1]=(_Float16)x0[1];  r[2]=(_Float16)x0[2];  r[3]=(_Float16)x0[3];
  r[4]=(_Float16)x1[0];  r[5]=(_Float16)x1[1];  r[6]=(_Float16)x1[2];  r[7]=(_Float16)x1[3];
  r[8]=(_Float16)x2[0];  r[9]=(_Float16)x2[1];  r[10]=(_Float16)x2[2]; r[11]=(_Float16)x2[3];
  r[12]=(_Float16)x3[0]; r[13]=(_Float16)x3[1]; r[14]=(_Float16)x3[2]; r[15]=(_Float16)x3[3];
  return r;
}

// A fragment from f16 memory (two 16B chunks: p[0..7], p[8..15]).
static __device__ __forceinline__ v16h a_frag_from_f16(const _Float16* p) {
  v8h lo = *(const v8h*)(p);
  v8h hi = *(const v8h*)(p + 8);
  v16h r;
#pragma unroll
  for (int i = 0; i < 8; ++i) { r[i] = lo[i]; r[i + 8] = hi[i]; }
  return r;
}

// ---- f32 -> f16 bulk convert (for weights) --------------------------------
__global__ __launch_bounds__(256)
void f32_to_f16_kernel(const float* __restrict__ in, _Float16* __restrict__ out,
                       int n8) {
  const int i = blockIdx.x * blockDim.x + threadIdx.x;  // 8 elements per thread
  if (i >= n8) return;
  const f32x4 x0 = *(const f32x4*)(in + (size_t)i * 8);
  const f32x4 x1 = *(const f32x4*)(in + (size_t)i * 8 + 4);
  v8h r;
  r[0]=(_Float16)x0[0]; r[1]=(_Float16)x0[1]; r[2]=(_Float16)x0[2]; r[3]=(_Float16)x0[3];
  r[4]=(_Float16)x1[0]; r[5]=(_Float16)x1[1]; r[6]=(_Float16)x1[2]; r[7]=(_Float16)x1[3];
  *(v8h*)(out + (size_t)i * 8) = r;
}

// ---- projection GEMM: dst = concat(src0,src1) @ W16^T + bias (f16 out) ----
// One wave computes a 32x64 output tile (2 M-subtiles x 4 N-subtiles = 8 WMMA
// per k-step); block = 8 waves stacked in M (256 rows x 64 cols).
// grid.x = rows/256, grid.y = DMODEL/64. transposed=1 stores [B][H][DK][NLT].
__global__ __launch_bounds__(256)
void proj_kernel(const float* __restrict__ src0, const float* __restrict__ src1,
                 int L0, int Ltot,
                 const _Float16* __restrict__ W16,   // [1024,1024] f16 row-major
                 const float* __restrict__ bias,
                 _Float16* __restrict__ dst, int transposed) {
  const int lane = threadIdx.x & 31;
  const int wave = threadIdx.x >> 5;
  const int lr   = lane & 15;       // column (B/C/D) or row (A) selector
  const int hi   = lane >> 4;       // half-wave selector
  const int L1   = Ltot - L0;

  const int rowbase = (blockIdx.x * 8 + wave) * 32;
  const int nbase   = blockIdx.y * 64;

  // per-lane A row pointers for the two 16-row subtiles (handles the concat)
  const float* arow[2];
#pragma unroll
  for (int mt = 0; mt < 2; ++mt) {
    const int grow = rowbase + mt * 16 + lr;
    const int b    = grow / Ltot;
    const int l    = grow - b * Ltot;
    arow[mt] = (l < L0)
        ? (src0 + ((size_t)b * L0 + l) * DMODEL)
        : (src1 + ((size_t)b * L1 + (l - L0)) * DMODEL);
  }

  // B-fragment base: lane's weight row (output column nbase+nt*16+lr)
  const _Float16* wrow = W16 + (size_t)(nbase + lr) * DMODEL;

  v8f acc[2][4] = {};

#pragma unroll 2
  for (int kb = 0; kb < DMODEL; kb += 32) {
    const v16h a0 = a_frag_from_f32(arow[0] + kb + hi * 16);
    const v16h a1 = a_frag_from_f32(arow[1] + kb + hi * 16);
    // four independent 32B B-fragment loads (clause-friendly)
    const v16h bf0 = *(const v16h*)(wrow + (size_t)0  * 16 * DMODEL + kb + hi * 16);
    const v16h bf1 = *(const v16h*)(wrow + (size_t)1  * 16 * DMODEL + kb + hi * 16);
    const v16h bf2 = *(const v16h*)(wrow + (size_t)2  * 16 * DMODEL + kb + hi * 16);
    const v16h bf3 = *(const v16h*)(wrow + (size_t)3  * 16 * DMODEL + kb + hi * 16);

    acc[0][0] = __builtin_amdgcn_wmma_f32_16x16x32_f16(false, a0, false, bf0, (short)0, acc[0][0], false, false);
    acc[1][0] = __builtin_amdgcn_wmma_f32_16x16x32_f16(false, a1, false, bf0, (short)0, acc[1][0], false, false);
    acc[0][1] = __builtin_amdgcn_wmma_f32_16x16x32_f16(false, a0, false, bf1, (short)0, acc[0][1], false, false);
    acc[1][1] = __builtin_amdgcn_wmma_f32_16x16x32_f16(false, a1, false, bf1, (short)0, acc[1][1], false, false);
    acc[0][2] = __builtin_amdgcn_wmma_f32_16x16x32_f16(false, a0, false, bf2, (short)0, acc[0][2], false, false);
    acc[1][2] = __builtin_amdgcn_wmma_f32_16x16x32_f16(false, a1, false, bf2, (short)0, acc[1][2], false, false);
    acc[0][3] = __builtin_amdgcn_wmma_f32_16x16x32_f16(false, a0, false, bf3, (short)0, acc[0][3], false, false);
    acc[1][3] = __builtin_amdgcn_wmma_f32_16x16x32_f16(false, a1, false, bf3, (short)0, acc[1][3], false, false);
  }

  // epilogue: + bias, convert to f16, store
#pragma unroll
  for (int mt = 0; mt < 2; ++mt) {
#pragma unroll
    for (int nt = 0; nt < 4; ++nt) {
      const int n = nbase + nt * 16 + lr;
      const float bv = bias[n];
#pragma unroll
      for (int v = 0; v < 8; ++v) {
        const int m = rowbase + mt * 16 + hi * 8 + v;   // global row
        const float val = acc[mt][nt][v] + bv;
        if (!transposed) {
          dst[(size_t)m * DMODEL + n] = (_Float16)val;
        } else {
          const int bb = m / Ltot, ll = m - bb * Ltot;
          const int h = n >> 6, d = n & 63;
          dst[(((size_t)bb * NHEADS + h) * DKH + d) * NLT + ll] = (_Float16)val;
        }
      }
    }
  }
}

// ---- fused attention per (b,h) --------------------------------------------
// block = 64 threads (2 waves); wave w owns query rows [16w, 16w+16).
__global__ __launch_bounds__(64)
void attn_kernel(const _Float16* __restrict__ Qh,   // [B,32,1024]
                 const _Float16* __restrict__ Kh,   // [B,160,1024]
                 const _Float16* __restrict__ VhT,  // [B,H,64,160]
                 const float* __restrict__ title_mask, // [B,32]
                 const float* __restrict__ body_mask,  // [B,32,128]
                 float* __restrict__ out) {            // [B,32,1024]
  __shared__ _Float16 Elds[2][16][176];  // exp'd scores, f16, padded rows
  __shared__ float    denom[2][16];

  const int bh = blockIdx.x;
  const int b  = bh >> 4;
  const int h  = bh & 15;
  const int lane = threadIdx.x & 31;
  const int wave = threadIdx.x >> 5;
  const int lr   = lane & 15;
  const int hi   = lane >> 4;
  const int mbase = wave * 16;

  // Q A-fragments for both K-steps (d 0..31, 32..63), resident all phase 1
  const _Float16* qrow =
      Qh + ((size_t)b * NLQ + mbase + lr) * DMODEL + h * DKH;
  const v16h aq0 = a_frag_from_f16(qrow + 0  + hi * 16);
  const v16h aq1 = a_frag_from_f16(qrow + 32 + hi * 16);

  float rs[8];
#pragma unroll
  for (int v = 0; v < 8; ++v) rs[v] = 0.0f;

  // -------- phase 1: scores -> exp*mask -> LDS (f16) + row-sum in regs -----
  for (int nt = 0; nt < 10; ++nt) {
    const int key = nt * 16 + lr;  // this lane's key column
    const _Float16* krow =
        Kh + ((size_t)b * NLT + key) * DMODEL + h * DKH;
    const v16h bk0 = *(const v16h*)(krow + 0  + hi * 16);
    const v16h bk1 = *(const v16h*)(krow + 32 + hi * 16);

    v8f s = {};
    s = __builtin_amdgcn_wmma_f32_16x16x32_f16(false, aq0, false, bk0,
                                               (short)0, s, false, false);
    s = __builtin_amdgcn_wmma_f32_16x16x32_f16(false, aq1, false, bk1,
                                               (short)0, s, false, false);

#pragma unroll
    for (int v = 0; v < 8; ++v) {
      const int ml = hi * 8 + v;        // local query row (0..15)
      const int q  = mbase + ml;        // query row (0..31)
      const float mrow = title_mask[(size_t)b * NLQ + q];
      const float mcol = (key < NLK)
          ? body_mask[((size_t)b * NLQ + q) * NLK + key]
          : title_mask[(size_t)b * NLQ + (key - NLK)];
      const float e = __expf(s[v] * 0.125f) * mcol * mrow;
      rs[v] += e;
      Elds[wave][ml][key] = (_Float16)e;
    }
  }

  // row-sum reduce across the 16 key-lanes (xor <= 8 stays in each half-wave)
#pragma unroll
  for (int v = 0; v < 8; ++v) {
    float r = rs[v];
    r += __shfl_xor(r, 1);
    r += __shfl_xor(r, 2);
    r += __shfl_xor(r, 4);
    r += __shfl_xor(r, 8);
    if (lr == 0) denom[wave][hi * 8 + v] = r;
  }
  __syncthreads();

  // -------- phase 2: ctx = E @ Vh, scaled by 1/(denom+1e-8) ----------------
  v8f acc[4] = {};
  for (int kb = 0; kb < NLT; kb += 32) {
    const v16h ae = a_frag_from_f16(&Elds[wave][lr][kb + hi * 16]);
#pragma unroll
    for (int nt = 0; nt < 4; ++nt) {
      const int d = nt * 16 + lr;  // this lane's output feature column
      const _Float16* vrow =
          VhT + (((size_t)b * NHEADS + h) * DKH + d) * NLT + kb + hi * 16;
      const v16h bv = *(const v16h*)vrow;
      acc[nt] = __builtin_amdgcn_wmma_f32_16x16x32_f16(
          false, ae, false, bv, (short)0, acc[nt], false, false);
    }
  }

#pragma unroll
  for (int nt = 0; nt < 4; ++nt) {
    const int d = nt * 16 + lr;
#pragma unroll
    for (int v = 0; v < 8; ++v) {
      const int ml = hi * 8 + v;
      const int q  = mbase + ml;
      const float inv = 1.0f / (denom[wave][ml] + 1e-8f);
      out[((size_t)b * NLQ + q) * DMODEL + h * DKH + d] = acc[nt][v] * inv;
    }
  }
}

// ---- host-side launch ------------------------------------------------------
extern "C" void kernel_launch(void* const* d_in, const int* in_sizes, int n_in,
                              void* d_out, int out_size, void* d_ws, size_t ws_size,
                              hipStream_t stream) {
  (void)in_sizes; (void)n_in; (void)out_size; (void)ws_size;

  const float* Q_seq      = (const float*)d_in[0];  // [B,32,1024]
  const float* K_seq      = (const float*)d_in[1];  // [B,128,1024]
  const float* V_seq      = (const float*)d_in[2];  // [B,128,1024]
  const float* title_mask = (const float*)d_in[3];  // [B,32]
  const float* body_mask  = (const float*)d_in[4];  // [B,32,128]
  const float* Wq         = (const float*)d_in[5];  // [1024,1024]
  const float* bq         = (const float*)d_in[6];
  const float* Wk         = (const float*)d_in[7];
  const float* bk         = (const float*)d_in[8];
  const float* Wv         = (const float*)d_in[9];
  const float* bv         = (const float*)d_in[10];
  float* out = (float*)d_out;

  const size_t qh_elems = (size_t)NB * NLQ * DMODEL;   //  4,194,304 f16
  const size_t kh_elems = (size_t)NB * NLT * DMODEL;   // 20,971,520 f16
  const size_t w_elems  = (size_t)DMODEL * DMODEL;     //  1,048,576 f16
  _Float16* Qh   = (_Float16*)d_ws;
  _Float16* Kh   = Qh + qh_elems;
  _Float16* VhT  = Kh + kh_elems;
  _Float16* Wq16 = VhT + kh_elems;
  _Float16* Wk16 = Wq16 + w_elems;
  _Float16* Wv16 = Wk16 + w_elems;   // total 98,566,144 bytes of d_ws

  // Convert weights to f16 once (3 x 1M elements, 8 per thread)
  const int n8 = (int)(w_elems / 8);
  f32_to_f16_kernel<<<(n8 + 255) / 256, 256, 0, stream>>>(Wq, Wq16, n8);
  f32_to_f16_kernel<<<(n8 + 255) / 256, 256, 0, stream>>>(Wk, Wk16, n8);
  f32_to_f16_kernel<<<(n8 + 255) / 256, 256, 0, stream>>>(Wv, Wv16, n8);

  // Q projection: rows = 128*32 = 4096 (256 rows per block)
  proj_kernel<<<dim3(4096 / 256, DMODEL / 64), 256, 0, stream>>>(
      Q_seq, Q_seq, NLQ, NLQ, Wq16, bq, Qh, 0);

  // K projection over concat(K_seq, Q_seq): rows = 128*160 = 20480
  proj_kernel<<<dim3(20480 / 256, DMODEL / 64), 256, 0, stream>>>(
      K_seq, Q_seq, NLK, NLT, Wk16, bk, Kh, 0);

  // V projection, stored transposed per head [B][H][64][160]
  proj_kernel<<<dim3(20480 / 256, DMODEL / 64), 256, 0, stream>>>(
      V_seq, Q_seq, NLK, NLT, Wv16, bv, VhT, 1);

  // fused attention: one block per (b, h)
  attn_kernel<<<NB * NHEADS, 64, 0, stream>>>(
      Qh, Kh, VhT, title_mask, body_mask, out);
}
